// Attention_50362786513376
// MI455X (gfx1250) — matrix operand order repaired and checked by reading, not compiled, not run on hardware
//
#include <hip/hip_runtime.h>
#include <math.h>

typedef __attribute__((ext_vector_type(16))) _Float16 v16h;
typedef __attribute__((ext_vector_type(8)))  _Float16 v8h;
typedef __attribute__((ext_vector_type(8)))  float    v8f;

#define HEADS 8
#define DKEY  64
#define NSEQ  2048
#define DIM   512
#define BATCH 4
#define QSCALE 0.125f /* 64^-0.5 */

static __device__ __forceinline__ v8f wmma_f16(v16h a, v16h b, v8f c) {
  // D(16x16 f32) = A(16x32 f16) x B(32x16 f16) + C
  return __builtin_amdgcn_wmma_f32_16x16x32_f16(false, a, false, b, (short)0, c,
                                                false, false);
}

static __device__ __forceinline__ v16h cat8(v8h a, v8h b) {
  return __builtin_shufflevector(a, b, 0, 1, 2, 3, 4, 5, 6, 7, 8, 9, 10, 11, 12,
                                 13, 14, 15);
}

// Native CDNA5 tanh transcendental; s_nop covers the TRANS result hazard
// (compiler cannot schedule around opaque asm).
static __device__ __forceinline__ float fast_tanh(float x) {
  float r;
  asm("v_tanh_f32 %0, %1\n\ts_nop 1" : "=v"(r) : "v"(x));
  return r;
}

// -------------------------------------------------------------------------
// f32 -> f16 streaming pack (8 elements/thread, b128 in / b128 out)
// -------------------------------------------------------------------------
__global__ __launch_bounds__(256) void pack_f16(const float* __restrict__ src,
                                                _Float16* __restrict__ dst,
                                                int n8) {
  for (int i = blockIdx.x * blockDim.x + threadIdx.x; i < n8;
       i += gridDim.x * blockDim.x) {
    const float4 a = ((const float4*)src)[2 * (size_t)i];
    const float4 b = ((const float4*)src)[2 * (size_t)i + 1];
    v8h o;
    o[0] = (_Float16)a.x; o[1] = (_Float16)a.y;
    o[2] = (_Float16)a.z; o[3] = (_Float16)a.w;
    o[4] = (_Float16)b.x; o[5] = (_Float16)b.y;
    o[6] = (_Float16)b.z; o[7] = (_Float16)b.w;
    ((v8h*)dst)[i] = o;
  }
}

// -------------------------------------------------------------------------
// V^T pack: Vt[b,h][d][n] = (f16) x[b][n][h*64+d]
// -------------------------------------------------------------------------
__global__ __launch_bounds__(256) void pack_v(const float* __restrict__ x,
                                              _Float16* __restrict__ Vt) {
  const size_t total = (size_t)BATCH * HEADS * DKEY * NSEQ;
  for (size_t i = (size_t)blockIdx.x * blockDim.x + threadIdx.x; i < total;
       i += (size_t)gridDim.x * blockDim.x) {
    const size_t nn = i & (NSEQ - 1);
    const size_t t  = i >> 11;
    const size_t d  = t & (DKEY - 1);
    const size_t bh = t >> 6;
    const size_t h  = bh & (HEADS - 1);
    const size_t bb = bh >> 3;
    Vt[i] = (_Float16)x[(bb * NSEQ + nn) * DIM + h * DKEY + d];
  }
}

// -------------------------------------------------------------------------
// Pass 1: Q = scale * x Wq^T, K = x Wk^T from pre-packed f16 operands.
// One wave per 16x16 output tile; x A-fragment shared by Q and K WMMAs.
// -------------------------------------------------------------------------
__global__ __launch_bounds__(128) void proj_qk(const _Float16* __restrict__ x16,
                                               const _Float16* __restrict__ wq16,
                                               const _Float16* __restrict__ wk16,
                                               _Float16* __restrict__ Qo,
                                               _Float16* __restrict__ Ko) {
  const int lane = threadIdx.x & 31;
  const int wave = threadIdx.x >> 5;
  const int lo = lane & 15, hi = lane >> 4;
  const int i0 = (blockIdx.x * 4 + wave) * 16;  // flat row tile over b*n
  const int o0 = blockIdx.y * 16;               // output-column tile

  const _Float16* xrow = x16 + (size_t)(i0 + lo) * DIM + 16 * hi;
  const _Float16* wqr  = wq16 + (size_t)(o0 + lo) * DIM + 16 * hi;
  const _Float16* wkr  = wk16 + (size_t)(o0 + lo) * DIM + 16 * hi;

  v8f cq = {}; v8f ck = {};
  for (int k = 0; k < DIM; k += 32) {
    // A 16x32 f16: lane=M; elems contiguous K=16*hi+e
    const v16h a = *(const v16h*)(xrow + k);
    // B 32x16 f16: lane=N (K-half by hi); elems contiguous K=16*hi+e
    const v16h bq = *(const v16h*)(wqr + k);
    const v16h bk = *(const v16h*)(wkr + k);
    cq = wmma_f16(a, bq, cq);
    ck = wmma_f16(a, bk, ck);
  }

  // C layout: lane=N, VGPR r -> M = r + 8*hi. Scatter into [b,h,n,64] f16.
  const int ocol = o0 + lo;
  const int h = ocol >> 6, d = ocol & 63;
#pragma unroll
  for (int r = 0; r < 8; ++r) {
    const int g  = i0 + r + 8 * hi;  // flat row in [0, b*n)
    const int bb = g >> 11, nn = g & (NSEQ - 1);
    const size_t idx = ((size_t)(bb * HEADS + h) * NSEQ + nn) * DKEY + d;
    Qo[idx] = (_Float16)(cq[r] * QSCALE);
    Ko[idx] = (_Float16)ck[r];
  }
}

// -------------------------------------------------------------------------
// Pass 2: out = tanh(Q K^T) V, streamed over j in 32-column chunks.
// 4 waves/block; each wave owns a 32-row strip-pair of one (b,h) and
// reuses the K/V B-fragments across both 16-row strips: 16 WMMAs/chunk.
// -------------------------------------------------------------------------
__global__ __launch_bounds__(128) void attn_kernel(const _Float16* __restrict__ Q,
                                                   const _Float16* __restrict__ K,
                                                   const _Float16* __restrict__ Vt,
                                                   float* __restrict__ out) {
  __shared__ _Float16 sS[4][32 * 32];  // per-wave C->A transpose staging (2 KB)
  const int lane = threadIdx.x & 31;
  const int wave = threadIdx.x >> 5;
  const int lo = lane & 15, hi = lane >> 4;
  const int i0 = (blockIdx.x * 4 + wave) * 32;  // 32 rows per wave
  const int h = blockIdx.y, b = blockIdx.z;

  const _Float16* Qh = Q  + (size_t)(b * HEADS + h) * NSEQ * DKEY;
  const _Float16* Kh = K  + (size_t)(b * HEADS + h) * NSEQ * DKEY;
  const _Float16* Vh = Vt + (size_t)(b * HEADS + h) * DKEY * NSEQ;
  _Float16* myS = sS[wave];

  // Persistent Q tiles: 2 strips x (16x64) as two 16x32 A fragments each.
  v16h aq[2][2];
#pragma unroll
  for (int s = 0; s < 2; ++s) {
    const _Float16* qrow = Qh + (size_t)(i0 + 16 * s + lo) * DKEY + 16 * hi;
    aq[s][0] = *(const v16h*)(qrow);
    aq[s][1] = *(const v16h*)(qrow + 32);
  }

  v8f acc[2][4] = {{v8f{}, v8f{}, v8f{}, v8f{}}, {v8f{}, v8f{}, v8f{}, v8f{}}};

  for (int j = 0; j < NSEQ; j += 32) {
    // ---- S(32x32) = Qtiles . K[j:j+32]^T : 8 WMMAs, K frags shared ----
    v8f sc[2][2];
#pragma unroll
    for (int t2 = 0; t2 < 2; ++t2) {
      const _Float16* kr = Kh + (size_t)(j + 16 * t2 + lo) * DKEY + 16 * hi;
      const v16h bk0 = *(const v16h*)(kr);
      const v16h bk1 = *(const v16h*)(kr + 32);
#pragma unroll
      for (int s = 0; s < 2; ++s) {
        v8f c = {};
        c = wmma_f16(aq[s][0], bk0, c);
        c = wmma_f16(aq[s][1], bk1, c);
        sc[s][t2] = c;
      }
    }

    // ---- tanh + C-layout -> A-layout transpose through LDS ----
#pragma unroll
    for (int s = 0; s < 2; ++s)
#pragma unroll
      for (int t2 = 0; t2 < 2; ++t2)
#pragma unroll
        for (int r = 0; r < 8; ++r) {
          const int row = 16 * s + r + 8 * hi;
          myS[row * 32 + 16 * t2 + lo] = (_Float16)fast_tanh(sc[s][t2][r]);
        }
    asm volatile("s_wait_dscnt 0" ::: "memory");  // DS RAW fence within wave

    v16h ap[2];
#pragma unroll
    for (int s = 0; s < 2; ++s) {
      const _Float16* sr = myS + (16 * s + lo) * 32 + 16 * hi;
      ap[s] = *(const v16h*)(sr);
    }

    // ---- acc(32x64) += tanhS(32x32) . V[j:j+32][0:64] : 8 WMMAs ----
#pragma unroll
    for (int nt = 0; nt < 4; ++nt) {
      const _Float16* vr = Vh + (size_t)(16 * nt + lo) * NSEQ + j + 16 * hi;
      const v16h bv = *(const v16h*)(vr);
#pragma unroll
      for (int s = 0; s < 2; ++s) acc[s][nt] = wmma_f16(ap[s], bv, acc[s][nt]);
    }
  }

  // ---- epilogue: out[b, i0+row, h*64 + 16*nt + N] (f32) ----
#pragma unroll
  for (int s = 0; s < 2; ++s)
#pragma unroll
    for (int nt = 0; nt < 4; ++nt)
#pragma unroll
      for (int r = 0; r < 8; ++r) {
        const int row = i0 + 16 * s + r + 8 * hi;
        const int col = h * DKEY + 16 * nt + lo;
        out[((size_t)b * NSEQ + row) * DIM + col] = acc[s][nt][r];
      }
}

// -------------------------------------------------------------------------
extern "C" void kernel_launch(void* const* d_in, const int* in_sizes, int n_in,
                              void* d_out, int out_size, void* d_ws, size_t ws_size,
                              hipStream_t stream) {
  const float* x  = (const float*)d_in[0];
  const float* Wq = (const float*)d_in[1];
  const float* Wk = (const float*)d_in[2];
  float* out = (float*)d_out;

  const size_t E  = (size_t)BATCH * NSEQ * DIM;  // 4.19M (= b*h*n*64 too)
  const size_t WE = (size_t)DIM * DIM;           // 262144
  _Float16* Qws  = (_Float16*)d_ws;  // 8.4 MB
  _Float16* Kws  = Qws + E;          // 8.4 MB
  _Float16* Vws  = Kws + E;          // 8.4 MB
  _Float16* X16  = Vws + E;          // 8.4 MB
  _Float16* Wq16 = X16 + E;          // 0.5 MB
  _Float16* Wk16 = Wq16 + WE;        // 0.5 MB   (ws total ~34.6 MB)

  pack_f16<<<1024, 256, 0, stream>>>(x,  X16,  (int)(E / 8));
  pack_f16<<<256,  256, 0, stream>>>(Wq, Wq16, (int)(WE / 8));
  pack_f16<<<256,  256, 0, stream>>>(Wk, Wk16, (int)(WE / 8));
  pack_v  <<<1024, 256, 0, stream>>>(x, Vws);

  dim3 g1(BATCH * NSEQ / 64, DIM / 16, 1);  // 128 x 32 blocks, 4 waves each
  proj_qk<<<g1, 128, 0, stream>>>(X16, Wq16, Wk16, Qws, Kws);

  dim3 g2(NSEQ / 128, HEADS, BATCH);        // 16 x 8 x 4 blocks, 4 waves each
  attn_kernel<<<g2, 128, 0, stream>>>(Qws, Kws, Vws, out);
}
